// AutoConvDecoder_15006615732329
// MI455X (gfx1250) — compile-verified, matrix operand order checked by reading, not run on hardware
//
#include <hip/hip_runtime.h>

// ---------------------------------------------------------------------------
// CDNA5 (gfx1250) PixelCNN++ forward: every conv / GEMM is an implicit GEMM
// on v_wmma_f32_16x16x32_bf16 (wave32).  fp32 activations in memory; bf16
// fragments: A from vectorized float4 loads + hw cvt; B pre-transposed to
// bf16 [tap][Cout16][Ceff], async-staged to LDS once per block
// (global_load_async_to_lds_b128 + s_wait_asynccnt), read back as ds_load.
// All index math is shift/mask (every divisor in this net is a power of 2).
// ---------------------------------------------------------------------------

typedef __bf16 bf16_t;
typedef bf16_t v16bf __attribute__((ext_vector_type(16)));
typedef bf16_t v8bf  __attribute__((ext_vector_type(8)));
typedef float  v8f   __attribute__((ext_vector_type(8)));

#if defined(__has_builtin)
#if __has_builtin(__builtin_amdgcn_global_load_async_to_lds_b128) && \
    __has_builtin(__builtin_amdgcn_s_wait_asynccnt)
#define HAVE_ASYNC_LDS 1
#endif
#endif

// async-copy builtin expects int4-vector pointers in AS1 (global) / AS3 (LDS)
typedef int v4i_vs __attribute__((vector_size(16)));
typedef __attribute__((address_space(1))) v4i_vs* gv4ip;
typedef __attribute__((address_space(3))) v4i_vs* lv4ip;

__device__ __forceinline__ float elu1(float v) {
  return v > 0.f ? v : (__expf(v) - 1.f);
}

__device__ __forceinline__ bf16_t to_bf16_rne(float f) {
  unsigned u = __builtin_bit_cast(unsigned, f);
  unsigned r = (u + 0x7FFFu + ((u >> 16) & 1u)) >> 16;
  unsigned short s = (unsigned short)r;
  return __builtin_bit_cast(bf16_t, s);
}

// Vector A-chunk: 8 consecutive channels [c0, c0+8) at pixel `pix`.
// Chunks are 8-aligned; all concat segment boundaries are multiples of 32,
// so one chunk always comes from a single contiguous source run.
__device__ __forceinline__ v8bf load_a8(const float* __restrict__ x0,
                                        const float* __restrict__ x1,
                                        int C0, int C1, int Ct, int mode,
                                        long pix, int c0, bool valid) {
  v8bf r;
  if (!valid) {
#pragma unroll
    for (int i = 0; i < 8; ++i) r[i] = (bf16_t)0.f;
    return r;
  }
  bool neg = false;
  int cc = c0;
  if (mode == 2 && cc >= Ct) { cc -= Ct; neg = true; }
  const float4* p = (cc < C0)
      ? (const float4*)(x0 + pix * (long)C0 + cc)
      : (const float4*)(x1 + pix * (long)C1 + (cc - C0));
  const float4 q0 = p[0];
  const float4 q1 = p[1];
  float f[8] = {q0.x, q0.y, q0.z, q0.w, q1.x, q1.y, q1.z, q1.w};
#pragma unroll
  for (int i = 0; i < 8; ++i) {
    float v = neg ? -f[i] : f[i];
    if (mode != 0) v = elu1(v);
    r[i] = (bf16_t)v;                       // native v_cvt f32->bf16 (RNE)
  }
  return r;
}

// Generic implicit-GEMM conv / deconv on WMMA.
// M = 16 consecutive output pixels (same row), N = 16 out channels,
// K = kh*kw*Ceff.  Fat path when Ceff % 32 == 0; stem path (Ceff==2,C0==1,
// mode==2, Ktot<=12) otherwise.
__global__ void conv_wmma_kernel(
    const float* __restrict__ x0, const float* __restrict__ x1,
    int C0, int C1, int mode,
    const float* __restrict__ w,        // fp32 HWIO (stem path only)
    const bf16_t* __restrict__ wt,      // bf16 [tap][Cout16][Ceff] (fat path)
    const float* __restrict__ bias,
    float* __restrict__ y, int accumulate,
    int Bn, int IH, int IW, int lgOH, int lgOW, int lgNT,
    int KH, int KW, int padtop, int padleft, int stride,
    int Cout, int deconv)
{
  __shared__ __align__(16) bf16_t lw[24576];      // 48 KB weight stage

  const int lane = threadIdx.x & 31;
  const int wave = threadIdx.x >> 5;
  const int WPB  = blockDim.x >> 5;

  const int Ct   = C0 + C1;
  const int Ceff = (mode == 2) ? 2 * Ct : Ct;
  const int Ktot = KH * KW * Ceff;
  const int NT   = 1 << lgNT;
  const int fat  = ((Ceff & 31) == 0);
  const int Cout16 = (Cout + 15) & ~15;

  // ---- cooperative weight staging into LDS (fat path, whole block) --------
  if (fat) {
    const int WB = KH * KW * Cout16 * Ceff * 2;   // bytes, <= 49152
#ifdef HAVE_ASYNC_LDS
    for (int ofs = (int)threadIdx.x * 16; ofs < WB; ofs += (int)blockDim.x * 16) {
      __builtin_amdgcn_global_load_async_to_lds_b128(
          (gv4ip)(void*)((const char*)wt + ofs),
          (lv4ip)(void*)((char*)lw + ofs), 0, 0);
    }
    __builtin_amdgcn_s_wait_asynccnt(0);          // this wave's copies landed
#else
    for (int ofs = (int)threadIdx.x * 16; ofs < WB; ofs += (int)blockDim.x * 16)
      *(uint4*)((char*)lw + ofs) = *(const uint4*)((const char*)wt + ofs);
#endif
    __syncthreads();                              // all waves' copies visible
  }

  const long tiles_total = ((long)Bn << (lgOH + lgOW + lgNT)) >> 4;
  long tile = (long)blockIdx.x * WPB + wave;
  if (tile >= tiles_total) return;

  const int  ntile = (int)tile & (NT - 1);
  const long mbase = (tile >> lgNT) << 4;
  const int  ow0   = (int)mbase & ((1 << lgOW) - 1);   // multiple of 16
  const long t     = mbase >> lgOW;
  const int  oh    = (int)t & ((1 << lgOH) - 1);
  const int  b     = (int)(t >> lgOH);

  const int klo = (lane & 16) ? 8 : 0;            // half-wave K sub-block
  const int mA  = lane & 15;                      // A row (pixel in tile)
  const int nB  = (ntile << 4) + (lane & 15);     // B / D column (out channel)
  const int owA = ow0 + mA;

  v8f acc = {0.f, 0.f, 0.f, 0.f, 0.f, 0.f, 0.f, 0.f};

  if (fat) {
    // ------------------- fat path: vectorized, division-free ---------------
    int tap = 0, cb = 0, khv = 0, kwv = 0;
    bool avalid = false;
    long apix = 0;
    v8f acc0 = {0.f, 0.f, 0.f, 0.f, 0.f, 0.f, 0.f, 0.f};
    v8f acc1 = {0.f, 0.f, 0.f, 0.f, 0.f, 0.f, 0.f, 0.f};
    int par = 0;

    for (int kb = 0; kb < Ktot; kb += 32) {
      if (cb == 0) {                      // entering a new (kh,kw) tap
        int hin, win;
        bool valid;
        if (!deconv) {
          hin = oh * stride + khv - padtop;
          win = owA * stride + kwv - padleft;
          valid = (hin >= 0) & (hin < IH) & (win >= 0) & (win < IW);
        } else {                          // stride-2 conv_transpose tap
          const int hn = oh + khv - padtop;
          const int wn = owA + kwv - padleft;
          valid = (hn >= 0) & (wn >= 0) & !(hn & 1) & !(wn & 1);
          hin = hn >> 1; win = wn >> 1;
          valid = valid & (hin < IH) & (win < IW);
        }
        avalid = valid;
        apix = ((long)b * IH + hin) * IW + win;
        if (valid)                        // gfx1250 global_prefetch_b8
          __builtin_prefetch(x0 + apix * (long)C0, 0, 1);
      }
      const int c0a = cb + klo;
      // A fragment: channels [c0a, c0a+8) and [c0a+16, c0a+24)
      const v8bf a_lo = load_a8(x0, x1, C0, C1, Ct, mode, apix, c0a, avalid);
      const v8bf a_hi = load_a8(x0, x1, C0, C1, Ct, mode, apix, c0a + 16, avalid);
      const v16bf afrag = __builtin_shufflevector(
          a_lo, a_hi, 0, 1, 2, 3, 4, 5, 6, 7, 8, 9, 10, 11, 12, 13, 14, 15);
      // B fragment: two contiguous 16B bf16 runs from LDS (ds_load_b128)
      const bf16_t* wr = lw + ((long)tap * Cout16 + nB) * Ceff + c0a;
      const v8bf b_lo = *(const v8bf*)wr;
      const v8bf b_hi = *(const v8bf*)(wr + 16);
      const v16bf bfrag = __builtin_shufflevector(
          b_lo, b_hi, 0, 1, 2, 3, 4, 5, 6, 7, 8, 9, 10, 11, 12, 13, 14, 15);

      if (par == 0)
        acc0 = __builtin_amdgcn_wmma_f32_16x16x32_bf16(
            false, afrag, false, bfrag, (short)0, acc0, false, false);
      else
        acc1 = __builtin_amdgcn_wmma_f32_16x16x32_bf16(
            false, afrag, false, bfrag, (short)0, acc1, false, false);
      par ^= 1;

      cb += 32;
      if (cb == Ceff) {
        cb = 0; ++tap;
        if (++kwv == KW) { kwv = 0; ++khv; }
      }
    }
#pragma unroll
    for (int v = 0; v < 8; ++v) acc[v] = acc0[v] + acc1[v];
  } else {
    // ---- stem path: C0==1, mode==2, Ceff==2, Ktot<=12, one K block --------
    v16bf afrag, bfrag;
#pragma unroll
    for (int e = 0; e < 16; ++e) {
      const int k = klo + (e < 8 ? e : e + 8);        // documented K map
      float av = 0.f, bv = 0.f;
      if (k < Ktot) {
        const int neg = k & 1;                        // channel within Ceff=2
        const int kc  = k >> 1;                       // tap index kh*KW+kw
        const int khv = (kc >= KW) + (kc >= 2 * KW);  // taps <= 6
        const int kwv = kc - khv * KW;
        const int hin = oh + khv - padtop;            // stride==1 in stem
        const int win = owA + kwv - padleft;
        if ((hin >= 0) & (hin < IH) & (win >= 0) & (win < IW)) {
          const float xv = x0[((long)b * IH + hin) * IW + win];
          av = elu1(neg ? -xv : xv);
        }
        if (nB < Cout)
          bv = w[((long)kc * 2 + neg) * Cout + nB];
      }
      afrag[e] = to_bf16_rne(av);
      bfrag[e] = to_bf16_rne(bv);
    }
    acc = __builtin_amdgcn_wmma_f32_16x16x32_bf16(
        false, afrag, false, bfrag, (short)0, acc, false, false);
  }

  if (nB < Cout) {
    const float bs = bias ? bias[nB] : 0.f;
    const int rofs = (lane & 16) ? 8 : 0;          // D-matrix row offset
#pragma unroll
    for (int v = 0; v < 8; ++v) {
      const int owD = ow0 + v + rofs;
      const long idx = (((((long)b << lgOH) | oh) << lgOW) | owD) * Cout + nB;
      const float val = acc[v] + bs;
      if (accumulate) y[idx] += val; else y[idx] = val;
    }
  }
}

// Weight prep: fp32 HWIO [T][Ceff][Cout] -> bf16 [T][Cout16][Ceff], zero-pad N.
__global__ void wprep_kernel(const float* __restrict__ w, bf16_t* __restrict__ o,
                             int lgCeff, int Cout, int lgCout16, long total) {
  long i = (long)blockIdx.x * blockDim.x + threadIdx.x;
  if (i >= total) return;
  const int  c  = (int)i & ((1 << lgCeff) - 1);
  const long tn = i >> lgCeff;
  const int  n  = (int)tn & ((1 << lgCout16) - 1);
  const int  tp = (int)(tn >> lgCout16);
  const float v = (n < Cout) ? w[(((long)tp << lgCeff) + c) * Cout + n] : 0.f;
  o[i] = to_bf16_rne(v);
}

__global__ void i2f_kernel(const int* __restrict__ in, float* __restrict__ out,
                           long n) {
  long i = (long)blockIdx.x * blockDim.x + threadIdx.x;
  if (i < n) out[i] = (float)in[i];
}

// x += a * sigmoid(g), with per-batch conditional bias folded in.  F == 32.
__global__ void gate_kernel(float* __restrict__ xio,
                            const float* __restrict__ c2,
                            const float* __restrict__ condb,
                            long total, int lgHW) {
  long i = (long)blockIdx.x * blockDim.x + threadIdx.x;
  if (i >= total) return;
  const int  f = (int)i & 31;
  const long p = i >> 5;                  // global pixel index b*HW + local
  const long b = p >> lgHW;
  const float a = c2[(p << 6) + f]      + condb[(b << 6) + f];
  const float g = c2[(p << 6) + 32 + f] + condb[(b << 6) + 32 + f];
  const float sg = 1.f / (1.f + __expf(-g));
  xio[i] = xio[i] + a * sg;
}

// Discretized mixture-of-logistics log-prob (M=2 mixtures, binary data).
__global__ void logprob_kernel(const float* __restrict__ net,
                               const int* __restrict__ xd,
                               float* __restrict__ out, int HW) {
  const int b = blockIdx.x;
  float acc = 0.f;
  for (int p = threadIdx.x; p < HW; p += blockDim.x) {
    const long o  = ((long)b * HW + p) * 6;
    const float xv = 2.f * (float)xd[(long)b * HW + p] - 1.f;   // exactly +-1
    float lp[2], lg[2];
#pragma unroll
    for (int m = 0; m < 2; ++m) {
      const float logit = net[o + m * 3 + 0];
      const float mean  = net[o + m * 3 + 1];
      const float logs  = net[o + m * 3 + 2];
      const float invs  = __expf(-fmaxf(logs, -7.f));
      const float plus_in = invs * (xv - mean + 1.f);
      const float min_in  = invs * (xv - mean - 1.f);
      const float z = (xv < 0.f) ? plus_in : -min_in;            // edge bins
      lp[m] = fminf(z, 0.f) - log1pf(__expf(-fabsf(z)));         // log sigmoid
      lg[m] = logit;
    }
    const float mxl = fmaxf(lg[0], lg[1]);
    const float lse = mxl + log1pf(__expf(fminf(lg[0], lg[1]) - mxl));
    const float t0 = lg[0] - lse + lp[0];
    const float t1 = lg[1] - lse + lp[1];
    const float m2 = fmaxf(t0, t1);
    acc += m2 + log1pf(__expf(fminf(t0, t1) - m2));
  }
  __shared__ float s[256];
  s[threadIdx.x] = acc;
  __syncthreads();
  for (int st = blockDim.x >> 1; st > 0; st >>= 1) {
    if ((int)threadIdx.x < st) s[threadIdx.x] += s[threadIdx.x + st];
    __syncthreads();
  }
  if (threadIdx.x == 0) out[b] = s[0];
}

// ---------------------------------------------------------------------------
// Host side
// ---------------------------------------------------------------------------

// JAX pytree leaf order: sorted dict keys, recursively.
enum {
  IN_LATENT = 0,
  IN_D0U_B1, IN_D0U_B2, IN_D0U_W1, IN_D0U_W2, IN_D0U_WC,                    // 1-5
  IN_D0UL_B1, IN_D0UL_B2, IN_D0UL_BA, IN_D0UL_W1, IN_D0UL_W2, IN_D0UL_WA,
  IN_D0UL_WC,                                                               // 6-12
  IN_D1U_B1, IN_D1U_B2, IN_D1U_W1, IN_D1U_W2, IN_D1U_WC,                    // 13-17
  IN_D1UL_B1, IN_D1UL_B2, IN_D1UL_BA, IN_D1UL_W1, IN_D1UL_W2, IN_D1UL_WA,
  IN_D1UL_WC,                                                               // 18-24
  IN_DSU_B, IN_DSU_W, IN_DSUL_B, IN_DSUL_W,                                 // 25-28
  IN_H0AB, IN_H0AW, IN_H0BB, IN_H0BW,                                       // 29-32
  IN_OUTB, IN_OUTW,                                                         // 33-34
  IN_U0U_B1, IN_U0U_B2, IN_U0U_BA, IN_U0U_W1, IN_U0U_W2, IN_U0U_WA,
  IN_U0U_WC,                                                                // 35-41
  IN_U0UL_B1, IN_U0UL_B2, IN_U0UL_BA, IN_U0UL_W1, IN_U0UL_W2, IN_U0UL_WA,
  IN_U0UL_WC,                                                               // 42-48
  IN_U1U_B1, IN_U1U_B2, IN_U1U_BA, IN_U1U_W1, IN_U1U_W2, IN_U1U_WA,
  IN_U1U_WC,                                                                // 49-55
  IN_U1UL_B1, IN_U1UL_B2, IN_U1UL_BA, IN_U1UL_W1, IN_U1UL_W2, IN_U1UL_WA,
  IN_U1UL_WC,                                                               // 56-62
  IN_USU_B, IN_USU_W, IN_USUL_B, IN_USUL_W,                                 // 63-66
  IN_V0B, IN_V0W,                                                           // 67-68
  IN_TRAIN                                                                  // 69
};

static inline int ilog2i(int v) { int l = 0; while ((1 << l) < v) ++l; return l; }

static inline void launch_conv(hipStream_t s,
    const float* x0, const float* x1, int C0, int C1, int mode,
    const float* w, const bf16_t* wt, const float* bias, float* y, int acc,
    int Bn, int IH, int IW, int OH, int OW,
    int KH, int KW, int pt, int pl, int stride, int Cout, int deconv)
{
  const int NT = (Cout + 15) >> 4;                  // 1, 2 or 4 (pow2)
  const long tiles = (((long)Bn * OH * OW) >> 4) * NT;
  const int wpb = 4;                                // 4 waves / block (wave32)
  const unsigned blocks = (unsigned)((tiles + wpb - 1) / wpb);
  conv_wmma_kernel<<<dim3(blocks), dim3(128), 0, s>>>(
      x0, x1, C0, C1, mode, w, wt, bias, y, acc,
      Bn, IH, IW, ilog2i(OH), ilog2i(OW), ilog2i(NT),
      KH, KW, pt, pl, stride, Cout, deconv);
}

extern "C" void kernel_launch(void* const* d_in, const int* in_sizes, int n_in,
                              void* d_out, int out_size, void* d_ws, size_t ws_size,
                              hipStream_t stream) {
  if (n_in < 70) return;
  (void)in_sizes; (void)out_size; (void)ws_size;

  const float* latent = (const float*)d_in[IN_LATENT];
  const int*   train  = (const int*)d_in[IN_TRAIN];
  #define PF(i) ((const float*)d_in[i])

  const int  Bn = 64;
  const long HWf = 64L * 64;        // 4096 pixels full res  (lg 12)
  const long HWh = 32L * 32;        // 1024 pixels half res  (lg 10)
  const long BHW  = Bn * HWf;       // 262144
  const long BHWh = Bn * HWh;       // 65536

  float* ws = (float*)d_ws;
  long off = 0;
  bf16_t* WT = (bf16_t*)(ws + off); off += 16384;  // prepped weights (<=64KB)
  float* XF  = ws + off; off += BHW;               // input as float
  float* CB  = ws + off; off += (long)Bn * 64;     // cond bias [B,2F]
  float* NET = ws + off; off += BHW * 6;           // final conv out
  float* A   = ws + off; off += BHW * 32;          // u  (full res)  -> u_s0
  float* Bu  = ws + off; off += BHW * 32;          // ul (full res)  -> ul_s0
  float* D   = ws + off; off += BHW * 32;          // u  (up pass)
  float* E   = ws + off; off += BHW * 32;          // ul (up pass)
  float* T1  = ws + off; off += BHW * 32;          // scratch, F ch
  float* T2  = ws + off; off += BHW * 64;          // scratch, 2F ch
  float* AH  = ws + off; off += BHWh * 32;         // u  half res == u_s1
  float* BH  = ws + off; off += BHWh * 32;         // ul half res == ul_s1

  i2f_kernel<<<(unsigned)((BHW + 255) / 256), 256, 0, stream>>>(train, XF, BHW);

  auto prep = [&](int widx, int T, int Ceff, int Cout) {
    const int Cout16 = (Cout + 15) & ~15;
    const long total = (long)T * Cout16 * Ceff;
    wprep_kernel<<<(unsigned)((total + 255) / 256), 256, 0, stream>>>(
        PF(widx), WT, ilog2i(Ceff), Cout, ilog2i(Cout16), total);
  };
  auto gate = [&](float* x, long HW, int lgHW) {
    const long total = Bn * HW * 32;
    gate_kernel<<<(unsigned)((total + 255) / 256), 256, 0, stream>>>(
        x, T2, CB, total, lgHW);
  };
  auto condmm = [&](int wc_idx) {  // CB[b,n] = latent[b,:] @ wc  (WMMA GEMM)
    prep(wc_idx, 1, 64, 64);
    launch_conv(stream, latent, nullptr, 64, 0, 0, PF(wc_idx), WT, nullptr,
                CB, 0, 1, 1, 64, 1, 64, 1, 1, 0, 0, 1, 64, 0);
  };

  // ---- initial causal stacks (shifts folded into pads): stem path ---------
  launch_conv(stream, XF, nullptr, 1, 0, 2, PF(IN_V0W), nullptr, PF(IN_V0B),
              A, 0, Bn, 64, 64, 64, 64, 2, 3, 2, 1, 1, 32, 0);
  launch_conv(stream, XF, nullptr, 1, 0, 2, PF(IN_H0AW), nullptr, PF(IN_H0AB),
              Bu, 0, Bn, 64, 64, 64, 64, 1, 3, 1, 1, 1, 32, 0);
  launch_conv(stream, XF, nullptr, 1, 0, 2, PF(IN_H0BW), nullptr, PF(IN_H0BB),
              Bu, 1, Bn, 64, 64, 64, 64, 2, 1, 1, 1, 1, 32, 0);

  // ---- d0u: gated resnet on u (ds 2x3) ------------------------------------
  prep(IN_D0U_W1, 6, 64, 32);
  launch_conv(stream, A, nullptr, 32, 0, 2, PF(IN_D0U_W1), WT, PF(IN_D0U_B1),
              T1, 0, Bn, 64, 64, 64, 64, 2, 3, 1, 1, 1, 32, 0);
  prep(IN_D0U_W2, 6, 64, 64);
  launch_conv(stream, T1, nullptr, 32, 0, 2, PF(IN_D0U_W2), WT, PF(IN_D0U_B2),
              T2, 0, Bn, 64, 64, 64, 64, 2, 3, 1, 1, 1, 64, 0);
  condmm(IN_D0U_WC);  gate(A, HWf, 12);

  // ---- d0ul: gated resnet on ul (dr 2x2), aux = u -------------------------
  prep(IN_D0UL_W1, 4, 64, 32);
  launch_conv(stream, Bu, nullptr, 32, 0, 2, PF(IN_D0UL_W1), WT, PF(IN_D0UL_B1),
              T1, 0, Bn, 64, 64, 64, 64, 2, 2, 1, 1, 1, 32, 0);
  prep(IN_D0UL_WA, 1, 64, 32);
  launch_conv(stream, A, nullptr, 32, 0, 2, PF(IN_D0UL_WA), WT, PF(IN_D0UL_BA),
              T1, 1, Bn, 64, 64, 64, 64, 1, 1, 0, 0, 1, 32, 0);
  prep(IN_D0UL_W2, 4, 64, 64);
  launch_conv(stream, T1, nullptr, 32, 0, 2, PF(IN_D0UL_W2), WT, PF(IN_D0UL_B2),
              T2, 0, Bn, 64, 64, 64, 64, 2, 2, 1, 1, 1, 64, 0);
  condmm(IN_D0UL_WC); gate(Bu, HWf, 12);
  // A == u_s0, Bu == ul_s0 from here on.

  // ---- stride-2 downsample (raw activations, mode 0) ----------------------
  prep(IN_DSU_W, 6, 32, 32);
  launch_conv(stream, A, nullptr, 32, 0, 0, PF(IN_DSU_W), WT, PF(IN_DSU_B),
              AH, 0, Bn, 64, 64, 32, 32, 2, 3, 1, 1, 2, 32, 0);
  prep(IN_DSUL_W, 4, 32, 32);
  launch_conv(stream, Bu, nullptr, 32, 0, 0, PF(IN_DSUL_W), WT, PF(IN_DSUL_B),
              BH, 0, Bn, 64, 64, 32, 32, 2, 2, 1, 1, 2, 32, 0);

  // ---- d1u -----------------------------------------------------------------
  prep(IN_D1U_W1, 6, 64, 32);
  launch_conv(stream, AH, nullptr, 32, 0, 2, PF(IN_D1U_W1), WT, PF(IN_D1U_B1),
              T1, 0, Bn, 32, 32, 32, 32, 2, 3, 1, 1, 1, 32, 0);
  prep(IN_D1U_W2, 6, 64, 64);
  launch_conv(stream, T1, nullptr, 32, 0, 2, PF(IN_D1U_W2), WT, PF(IN_D1U_B2),
              T2, 0, Bn, 32, 32, 32, 32, 2, 3, 1, 1, 1, 64, 0);
  condmm(IN_D1U_WC);  gate(AH, HWh, 10);

  // ---- d1ul, aux = u -------------------------------------------------------
  prep(IN_D1UL_W1, 4, 64, 32);
  launch_conv(stream, BH, nullptr, 32, 0, 2, PF(IN_D1UL_W1), WT, PF(IN_D1UL_B1),
              T1, 0, Bn, 32, 32, 32, 32, 2, 2, 1, 1, 1, 32, 0);
  prep(IN_D1UL_WA, 1, 64, 32);
  launch_conv(stream, AH, nullptr, 32, 0, 2, PF(IN_D1UL_WA), WT, PF(IN_D1UL_BA),
              T1, 1, Bn, 32, 32, 32, 32, 1, 1, 0, 0, 1, 32, 0);
  prep(IN_D1UL_W2, 4, 64, 64);
  launch_conv(stream, T1, nullptr, 32, 0, 2, PF(IN_D1UL_W2), WT, PF(IN_D1UL_B2),
              T2, 0, Bn, 32, 32, 32, 32, 2, 2, 1, 1, 1, 64, 0);
  condmm(IN_D1UL_WC); gate(BH, HWh, 10);
  // AH == u_s1, BH == ul_s1 (skip values intact until their gate).

  // ---- u1u: aux = u_s1 (same buffer, read-only until gate) ----------------
  prep(IN_U1U_W1, 6, 64, 32);
  launch_conv(stream, AH, nullptr, 32, 0, 2, PF(IN_U1U_W1), WT, PF(IN_U1U_B1),
              T1, 0, Bn, 32, 32, 32, 32, 2, 3, 1, 1, 1, 32, 0);
  prep(IN_U1U_WA, 1, 64, 32);
  launch_conv(stream, AH, nullptr, 32, 0, 2, PF(IN_U1U_WA), WT, PF(IN_U1U_BA),
              T1, 1, Bn, 32, 32, 32, 32, 1, 1, 0, 0, 1, 32, 0);
  prep(IN_U1U_W2, 6, 64, 64);
  launch_conv(stream, T1, nullptr, 32, 0, 2, PF(IN_U1U_W2), WT, PF(IN_U1U_B2),
              T2, 0, Bn, 32, 32, 32, 32, 2, 3, 1, 1, 1, 64, 0);
  condmm(IN_U1U_WC);  gate(AH, HWh, 10);

  // ---- u1ul: aux = concat([u_new, ul_s1])  (Ceff = 128) -------------------
  prep(IN_U1UL_W1, 4, 64, 32);
  launch_conv(stream, BH, nullptr, 32, 0, 2, PF(IN_U1UL_W1), WT, PF(IN_U1UL_B1),
              T1, 0, Bn, 32, 32, 32, 32, 2, 2, 1, 1, 1, 32, 0);
  prep(IN_U1UL_WA, 1, 128, 32);
  launch_conv(stream, AH, BH, 32, 32, 2, PF(IN_U1UL_WA), WT, PF(IN_U1UL_BA),
              T1, 1, Bn, 32, 32, 32, 32, 1, 1, 0, 0, 1, 32, 0);
  prep(IN_U1UL_W2, 4, 64, 64);
  launch_conv(stream, T1, nullptr, 32, 0, 2, PF(IN_U1UL_W2), WT, PF(IN_U1UL_B2),
              T2, 0, Bn, 32, 32, 32, 32, 2, 2, 1, 1, 1, 64, 0);
  condmm(IN_U1UL_WC); gate(BH, HWh, 10);

  // ---- stride-2 transposed convs back to full res -------------------------
  prep(IN_USU_W, 6, 32, 32);
  launch_conv(stream, AH, nullptr, 32, 0, 0, PF(IN_USU_W), WT, PF(IN_USU_B),
              D, 0, Bn, 32, 32, 64, 64, 2, 3, 1, 1, 1, 32, 1);
  prep(IN_USUL_W, 4, 32, 32);
  launch_conv(stream, BH, nullptr, 32, 0, 0, PF(IN_USUL_W), WT, PF(IN_USUL_B),
              E, 0, Bn, 32, 32, 64, 64, 2, 2, 1, 1, 1, 32, 1);

  // ---- u0u: aux = u_s0 (A) -------------------------------------------------
  prep(IN_U0U_W1, 6, 64, 32);
  launch_conv(stream, D, nullptr, 32, 0, 2, PF(IN_U0U_W1), WT, PF(IN_U0U_B1),
              T1, 0, Bn, 64, 64, 64, 64, 2, 3, 1, 1, 1, 32, 0);
  prep(IN_U0U_WA, 1, 64, 32);
  launch_conv(stream, A, nullptr, 32, 0, 2, PF(IN_U0U_WA), WT, PF(IN_U0U_BA),
              T1, 1, Bn, 64, 64, 64, 64, 1, 1, 0, 0, 1, 32, 0);
  prep(IN_U0U_W2, 6, 64, 64);
  launch_conv(stream, T1, nullptr, 32, 0, 2, PF(IN_U0U_W2), WT, PF(IN_U0U_B2),
              T2, 0, Bn, 64, 64, 64, 64, 2, 3, 1, 1, 1, 64, 0);
  condmm(IN_U0U_WC);  gate(D, HWf, 12);

  // ---- u0ul: aux = concat([u_new, ul_s0]) ---------------------------------
  prep(IN_U0UL_W1, 4, 64, 32);
  launch_conv(stream, E, nullptr, 32, 0, 2, PF(IN_U0UL_W1), WT, PF(IN_U0UL_B1),
              T1, 0, Bn, 64, 64, 64, 64, 2, 2, 1, 1, 1, 32, 0);
  prep(IN_U0UL_WA, 1, 128, 32);
  launch_conv(stream, D, Bu, 32, 32, 2, PF(IN_U0UL_WA), WT, PF(IN_U0UL_BA),
              T1, 1, Bn, 64, 64, 64, 64, 1, 1, 0, 0, 1, 32, 0);
  prep(IN_U0UL_W2, 4, 64, 64);
  launch_conv(stream, T1, nullptr, 32, 0, 2, PF(IN_U0UL_W2), WT, PF(IN_U0UL_B2),
              T2, 0, Bn, 64, 64, 64, 64, 2, 2, 1, 1, 1, 64, 0);
  condmm(IN_U0UL_WC); gate(E, HWf, 12);

  // ---- final 1x1 conv on elu(ul): 32 -> 6 (fat path, N padded to 16) ------
  prep(IN_OUTW, 1, 32, 6);
  launch_conv(stream, E, nullptr, 32, 0, 1, PF(IN_OUTW), WT, PF(IN_OUTB),
              NET, 0, Bn, 64, 64, 64, 64, 1, 1, 0, 0, 1, 6, 0);

  // ---- discretized mixture-of-logistics log-prob -> out[B] ----------------
  logprob_kernel<<<dim3(Bn), dim3(256), 0, stream>>>(
      NET, train, (float*)d_out, (int)HWf);
  #undef PF
}
